// NeuMissBlock_6347961663676
// MI455X (gfx1250) — compile-verified, matrix operand order checked by reading, not run on hardware
//
#include <hip/hip_runtime.h>

// ---------------------------------------------------------------------------
// NeuMiss block on gfx1250.  h_res = nan_to_num(x) - nm*mu;
// 10x: h = (h @ W^T) * nm + h_res.
// Rows evolve independently, so each WG keeps its 32-row state resident in
// LDS for all 10 steps (ping-pong h buffers), and streams a pre-swizzled
// bf16 copy of W (d_ws) through v_wmma_f32_16x16x32_bf16 with an explicit
// 2-deep software pipeline on the B-fragment stream.
// ---------------------------------------------------------------------------

typedef __attribute__((ext_vector_type(16))) __bf16 v16bf;
typedef __attribute__((ext_vector_type(8)))  float  v8f;

#define N_FEAT     1024
#define DEPTH      10
#define BLOCK_ROWS 32
#define NTHREADS   256

// padded LDS strides (elements)
#define HSTR   1040   // bf16 h   : 2080 B/row (32B multiple, bank-spread)
#define HRSTR  1028   // f32 h_res: 4112 B/row

__device__ __forceinline__ unsigned short f2bf(float f) {
    union { float f; unsigned u; } v; v.f = f;
    unsigned u = v.u;
    unsigned r = u + 0x7FFFu + ((u >> 16) & 1u);   // round-to-nearest-even
    return (unsigned short)(r >> 16);
}

// ---------------------------------------------------------------------------
// Pre-pass: f32 W[g,f] -> bf16 WMMA B-fragments in d_ws (2 MiB).
// Fragment id f = ctile*32 + kstep; lane l holds 16 contiguous bf16 of
// column g = ctile*16 + (l&15), k = kstep*32 + (l>>4)*16 + e.
// ---------------------------------------------------------------------------
__global__ void neumiss_wswizzle(const float* __restrict__ W,
                                 unsigned short* __restrict__ Wb) {
    const int t    = blockIdx.x * blockDim.x + threadIdx.x;  // 0 .. 65535
    const int lane = t & 31;
    const int s    = (t >> 5) & 31;
    const int c    = t >> 10;
    const int g    = c * 16 + (lane & 15);
    const int k0   = s * 32 + ((lane >> 4) << 4);
    const float* src = W + (size_t)g * N_FEAT + k0;
    unsigned short* dst = Wb + ((size_t)t << 4);
#pragma unroll
    for (int e = 0; e < 16; ++e) dst[e] = f2bf(src[e]);
}

// ---------------------------------------------------------------------------
// Main persistent-state kernel: one WG (8 wave32s) per 32-row block.
// ---------------------------------------------------------------------------
__global__ __launch_bounds__(NTHREADS, 1)
void neumiss_block_kernel(const float* __restrict__ x,
                          const float* __restrict__ mu,
                          const unsigned short* __restrict__ Wb,
                          float* __restrict__ out) {
    extern __shared__ char smem_raw[];
    unsigned short* h0_lds = (unsigned short*)smem_raw;                // 32*HSTR
    unsigned short* h1_lds = h0_lds + BLOCK_ROWS * HSTR;               // 32*HSTR
    float*          hr_lds = (float*)(h1_lds + BLOCK_ROWS * HSTR);     // 32*HRSTR
    unsigned*       nmb    = (unsigned*)(hr_lds + BLOCK_ROWS * HRSTR); // 32*32

    const int tid = threadIdx.x;
    const int r0  = blockIdx.x * BLOCK_ROWS;

    // ---- Phase 0: nm bitmask, h_res (f32), h0 = h_res (bf16) ---------------
#pragma unroll
    for (int j = 0; j < (BLOCK_ROWS * 32) / NTHREADS; ++j) {   // 4 iters
        const int idx  = tid + NTHREADS * j;    // 0..1023
        const int row  = idx >> 5;              // 0..31
        const int cg   = idx & 31;              // 32-col group
        const int col0 = cg << 5;
        unsigned bits = 0u;
#pragma unroll
        for (int v = 0; v < 8; ++v) {
            const float4 xv  = *(const float4*)(x + (size_t)(r0 + row) * N_FEAT
                                                  + col0 + v * 4);
            const float4 muv = *(const float4*)(mu + col0 + v * 4);
            const float xs[4] = {xv.x, xv.y, xv.z, xv.w};
            const float ms[4] = {muv.x, muv.y, muv.z, muv.w};
#pragma unroll
            for (int e = 0; e < 4; ++e) {
                const float xe   = xs[e];
                const bool  obs  = (xe == xe);              // !isnan
                const float hres = obs ? (xe - ms[e]) : 0.0f;
                const int   col  = col0 + v * 4 + e;
                bits |= ((unsigned)obs) << (v * 4 + e);
                hr_lds[row * HRSTR + col] = hres;
                h0_lds[row * HSTR  + col] = f2bf(hres);
            }
        }
        nmb[(row << 5) + cg] = bits;
    }
    __syncthreads();

    // ---- wave tiling -------------------------------------------------------
    const int lane  = tid & 31;
    const int w     = tid >> 5;            // 8 wave32s
    const int rtile = w >> 2;              // 0..1 : 16-row half
    const int wcol  = w & 3;               // 0..3 : 256-column group
    const int arow  = rtile * 16 + (lane & 15);   // A-fragment row
    const int ak    = (lane >> 4) << 4;           // A-fragment k sub-offset
    const int ccol  = lane & 15;                  // C-fragment column in tile
    const int mrow0 = rtile * 16 + ((lane >> 4) << 3);  // C-fragment row base

    // ---- depth loop --------------------------------------------------------
    for (int it = 0; it < DEPTH; ++it) {
        const bool last = (it == DEPTH - 1);
        const unsigned short* hc = (it & 1) ? h1_lds : h0_lds;  // read
        unsigned short*       hn = (it & 1) ? h0_lds : h1_lds;  // write

        const unsigned short* aptr = hc + arow * HSTR + ak;

#pragma unroll
        for (int g = 0; g < 2; ++g) {               // 2 groups of 8 tiles
            // per-lane W base for this group's first tile (ushort units)
            const unsigned short* wp =
                Wb + (size_t)lane * 16 + (size_t)((wcol * 16 + g * 8) * 16384);

            v8f acc[8];
#pragma unroll
            for (int tt = 0; tt < 8; ++tt)
                acc[tt] = (v8f){0.f, 0.f, 0.f, 0.f, 0.f, 0.f, 0.f, 0.f};

            // ---- software pipeline: 2-deep B rotation, 1-deep A ------------
            v16bf bbuf[2];
            bbuf[0] = *(const v16bf*)(wp);             // (tt=0, s=0)
            bbuf[1] = *(const v16bf*)(wp + 16384);     // (tt=1, s=0)
            v16bf acur = *(const v16bf*)(aptr);        // s=0

#pragma unroll 4
            for (int s = 0; s < 32; ++s) {             // K = 32 steps of 32
                const v16bf anext = *(const v16bf*)(aptr + (s + 1) * 32);
                const unsigned short* wps = wp + s * 512;
#pragma unroll
                for (int tt = 0; tt < 8; ++tt) {
                    // prefetch the fragment 2 WMMAs ahead (compile-time imm;
                    // the s=31 overshoot stays inside this group's 256 KB)
                    const int nim = (tt < 6) ? ((tt + 2) * 16384)
                                             : ((tt - 6) * 16384 + 512);
                    const v16bf bnext = *(const v16bf*)(wps + nim);
                    acc[tt] = __builtin_amdgcn_wmma_f32_16x16x32_bf16(
                        false, acur, false, bbuf[tt & 1], (short)0,
                        acc[tt], false, false);
                    bbuf[tt & 1] = bnext;
                }
                acur = anext;
            }

            // ---- fused epilogue: acc*nm + h_res, tile pairs share nm word --
#pragma unroll
            for (int tp = 0; tp < 4; ++tp) {
                const int c0   = wcol * 16 + g * 8 + tp * 2;  // even tile idx
                const int colw = c0 >> 1;                     // nm word index
                unsigned bw[8];
#pragma unroll
                for (int i = 0; i < 8; ++i)
                    bw[i] = nmb[((mrow0 + i) << 5) + colw];

#pragma unroll
                for (int sub = 0; sub < 2; ++sub) {
                    const int tt     = tp * 2 + sub;
                    const int col    = (c0 + sub) * 16 + ccol;
                    const int bitpos = sub * 16 + ccol;
#pragma unroll
                    for (int i = 0; i < 8; ++i) {
                        const int   m   = mrow0 + i;
                        const float nmv =
                            ((bw[i] >> bitpos) & 1u) ? 1.0f : 0.0f;
                        const float hr  = hr_lds[m * HRSTR + col];
                        const float v   = acc[tt][i] * nmv + hr;
                        if (last) {
                            out[(size_t)(r0 + m) * N_FEAT + col] = v;
                        } else {
                            hn[m * HSTR + col] = f2bf(v);
                        }
                    }
                }
            }
        }

        if (!last) __syncthreads();   // next iter reads hn, overwrites hc
    }
}

// ---------------------------------------------------------------------------
extern "C" void kernel_launch(void* const* d_in, const int* in_sizes, int n_in,
                              void* d_out, int out_size, void* d_ws, size_t ws_size,
                              hipStream_t stream) {
    const float* x  = (const float*)d_in[0];   // [16384,1024] f32 (NaN = missing)
    const float* mu = (const float*)d_in[1];   // [1024] f32
    const float* W  = (const float*)d_in[2];   // [1024,1024] f32
    float* out = (float*)d_out;                // [16384,1024] f32
    unsigned short* Wb = (unsigned short*)d_ws;  // 2 MiB bf16 swizzled W

    neumiss_wswizzle<<<(64 * 32 * 32) / NTHREADS, NTHREADS, 0, stream>>>(W, Wb);

    const size_t lds_bytes = (size_t)BLOCK_ROWS * HSTR * 2 * 2   // h ping-pong
                           + (size_t)BLOCK_ROWS * HRSTR * 4      // f32 h_res
                           + (size_t)BLOCK_ROWS * 32 * 4;        // nm bitmask
    neumiss_block_kernel<<<16384 / BLOCK_ROWS, NTHREADS, lds_bytes, stream>>>(
        x, mu, Wb, out);
}
